// TransformerBlock_67482526154829
// MI455X (gfx1250) — matrix operand order repaired and checked by reading, not compiled, not run on hardware
//
#include <hip/hip_runtime.h>
#include <hip/hip_bf16.h>

// ---------------------------------------------------------------------------
// Transformer block for MI455X (gfx1250): all GEMMs + attention on
// v_wmma_f32_16x16x32_bf16 (wave32). fp32 inputs are truncated to bf16 once
// (bf16 keeps fp32 exponent range; attention logits here reach O(1e3-1e4)).
// Tile fills use GLOBAL_LOAD_ASYNC_TO_LDS_B128 (ASYNCcnt) where the copy is
// contiguous, with LDS double-buffering in the GEMM to hide HBM latency.
// ---------------------------------------------------------------------------

#define DMODEL 2048
#define NHEADS 16
#define DHEAD  128
#define DFF    8192
#define SEQ    2048
#define QKVN   (3 * DMODEL)   // 6144 fused Q|K|V columns

typedef __attribute__((ext_vector_type(16))) __bf16        bf16x16;
typedef __attribute__((ext_vector_type(8)))  float         f32x8;
typedef __attribute__((ext_vector_type(4)))  unsigned int  u32x4;
typedef int v4i_async __attribute__((vector_size(16)));    // builtin's ptr pointee

union FragB16 {
    bf16x16 v;
    u32x4   q[2];
    unsigned short us[16];
};

__device__ __forceinline__ unsigned short f32_to_bf16u(float f) {
    unsigned int u = __builtin_bit_cast(unsigned int, f);
    unsigned int r = u + 0x7FFFu + ((u >> 16) & 1u);   // round-to-nearest-even
    return (unsigned short)(r >> 16);
}

// ---------------------------------------------------------------------------
// Async global->LDS 16-byte copy (CDNA5 GLOBAL_LOAD_ASYNC_TO_LDS_B128,
// tracked by ASYNCcnt). Falls back to a synchronous VGPR round-trip if the
// toolchain lacks the builtin.
// ---------------------------------------------------------------------------
#if defined(__AMDGCN__) &&                                              \
    __has_builtin(__builtin_amdgcn_global_load_async_to_lds_b128) &&    \
    __has_builtin(__builtin_amdgcn_s_wait_asynccnt)
#define USE_ASYNC_COPY 1
#else
#define USE_ASYNC_COPY 0
#endif

__device__ __forceinline__ void copy16_async(const unsigned short* g,
                                             unsigned short* l) {
#if USE_ASYNC_COPY
    __builtin_amdgcn_global_load_async_to_lds_b128(
        (v4i_async*)g, (v4i_async*)l, 0, 0);
#else
    *(u32x4*)l = *(const u32x4*)g;
#endif
}

template <int N>
__device__ __forceinline__ void async_fence() {
#if USE_ASYNC_COPY
    __builtin_amdgcn_s_wait_asynccnt(N);
#endif
}

// ---------------------------------------------------------------------------
// Elementwise f32 -> bf16
// ---------------------------------------------------------------------------
__global__ __launch_bounds__(256) void cvt_bf16_kernel(
    const float* __restrict__ in, unsigned short* __restrict__ out, int n) {
    int i = blockIdx.x * 256 + threadIdx.x;
    if (i < n) out[i] = f32_to_bf16u(in[i]);
}

// Pack wq/wk/wv [H, D, 128] into fused bf16 weight [D, 6144]:
// col n: part = n/2048 (Q,K,V), head = (n%2048)/128, k = n%128
__global__ __launch_bounds__(256) void pack_wqkv_kernel(
    const float* __restrict__ wq, const float* __restrict__ wk,
    const float* __restrict__ wv, unsigned short* __restrict__ out) {
    int idx = blockIdx.x * 256 + threadIdx.x;      // 2048*6144 total
    int d = idx / QKVN, n = idx % QKVN;
    int part = n >> 11, rem = n & 2047;
    int h = rem >> 7, kk = rem & 127;
    const float* w = (part == 0) ? wq : (part == 1) ? wk : wv;
    out[idx] = f32_to_bf16u(w[((size_t)h * DMODEL + d) * DHEAD + kk]);
}

// ---------------------------------------------------------------------------
// Tiled bf16 WMMA GEMM: C[M,N] = A[M,K] * B[K,N]  (row-major, f32 accumulate)
// Block: 256 threads (8 waves), 128x128 tile, K-step 32, double-buffered LDS.
// A tile fills are async (2 GLOBAL_LOAD_ASYNC_TO_LDS_B128 per wave per tile);
// async-load "done" returns in order, so s_wait_asynccnt<=2 after issuing the
// next tile guarantees the current tile's copies have landed.
// ---------------------------------------------------------------------------
template <bool HAS_BIAS, bool RELU, bool OUT_BF16>
__global__ __launch_bounds__(256) void gemm_bf16_wmma(
    const unsigned short* __restrict__ A, const unsigned short* __restrict__ B,
    const float* __restrict__ bias, float* __restrict__ Cf,
    unsigned short* __restrict__ Cb, int M, int N, int K) {
    __shared__ unsigned short At[2][128 * 40];   // [row][k], k-stride 40 (pad)
    __shared__ unsigned short Bt[2][128 * 40];   // transposed: [n][k]

    const int t = threadIdx.x;
    const int wave = t >> 5, lane = t & 31;
    const int ln = lane & 15, hh = lane >> 4;
    const int mBase = blockIdx.y * 128;
    const int nBase = blockIdx.x * 128;

    auto loadA = [&](int buf, int k0) {
#pragma unroll
        for (int r2 = 0; r2 < 2; ++r2) {
            int id = t + r2 * 256;
            int row = id >> 2, cc = (id & 3) * 8;
            copy16_async(A + (size_t)(mBase + row) * K + k0 + cc,
                         At[buf] + row * 40 + cc);
        }
    };
    auto loadB = [&](int buf, int k0) {
#pragma unroll
        for (int r2 = 0; r2 < 2; ++r2) {
            int id = t + r2 * 256;
            int kr = id >> 4, n0 = (id & 15) * 8;
            u32x4 v = *(const u32x4*)(B + (size_t)(k0 + kr) * N + nBase + n0);
            const unsigned short* pv = (const unsigned short*)&v;
#pragma unroll
            for (int j = 0; j < 8; ++j) Bt[buf][(n0 + j) * 40 + kr] = pv[j];
        }
    };

    f32x8 acc[8];
    const f32x8 zero = {0.f, 0.f, 0.f, 0.f, 0.f, 0.f, 0.f, 0.f};
#pragma unroll
    for (int i = 0; i < 8; ++i) acc[i] = zero;

    loadA(0, 0);
    loadB(0, 0);

    for (int k0 = 0; k0 < K; k0 += 32) {
        const int cur = (k0 >> 5) & 1;
        if (k0 + 32 < K) {
            loadA(cur ^ 1, k0 + 32);
            loadB(cur ^ 1, k0 + 32);
            async_fence<2>();   // current buffer's async copies complete
        } else {
            async_fence<0>();
        }
        __syncthreads();

        const unsigned short* Atc = At[cur];
        const unsigned short* Btc = Bt[cur];
        // A fragment: lane row M = 16*wave + ln; k = 16*(i>=8) + 8*hh + (i&7)
        FragB16 a;
        a.q[0] = *(const u32x4*)(Atc + (16 * wave + ln) * 40 + 8 * hh);
        a.q[1] = *(const u32x4*)(Atc + (16 * wave + ln) * 40 + 16 + 8 * hh);
#pragma unroll
        for (int nt = 0; nt < 8; ++nt) {
            // B fragment: lane col N = 16*nt + ln; k = 16*(i>=8) + 8*hh + (i&7)
            FragB16 b;
            b.q[0] = *(const u32x4*)(Btc + (nt * 16 + ln) * 40 + 8 * hh);
            b.q[1] = *(const u32x4*)(Btc + (nt * 16 + ln) * 40 + 16 + 8 * hh);
            acc[nt] = __builtin_amdgcn_wmma_f32_16x16x32_bf16(
                false, a.v, false, b.v, (short)0, acc[nt], false, false);
        }
        __syncthreads();   // readers done before next overwrite of buf[cur]
    }

    // Epilogue: C/D layout -> lane holds col N = 16*nt+ln, rows 8*hh + r
#pragma unroll
    for (int nt = 0; nt < 8; ++nt) {
        int col = nBase + nt * 16 + ln;
        float bv = HAS_BIAS ? bias[col] : 0.0f;
#pragma unroll
        for (int r = 0; r < 8; ++r) {
            int row = mBase + 16 * wave + 8 * hh + r;
            float v = acc[nt][r] + bv;
            if (RELU) v = fmaxf(v, 0.0f);
            if (OUT_BF16)
                Cb[(size_t)row * N + col] = f32_to_bf16u(v);
            else
                Cf[(size_t)row * N + col] = v;
        }
    }
}

// ---------------------------------------------------------------------------
// Flash attention: block = (head, 128 query rows), 8 waves of 32.
// Wave w owns q rows 16w..16w+15 and all 128 v columns. Online softmax.
// qkv: bf16 [SEQ, 6144] fused (Q|K|V, head-major 128 within each part).
// K tile fill is async and overlaps the V transpose-scatter work.
// ---------------------------------------------------------------------------
__global__ __launch_bounds__(256) void attn_flash_kernel(
    const unsigned short* __restrict__ qkv, unsigned short* __restrict__ heads) {
    __shared__ unsigned short Kt[64 * 136];       // [key][dk], stride 136
    __shared__ unsigned short Vt[128 * 72];       // transposed: [col][key]
    __shared__ unsigned short Pst[8 * 16 * 72];   // per-wave P [16][64] pad 72

    const int t = threadIdx.x;
    const int wave = t >> 5, lane = t & 31;
    const int ln = lane & 15, hh = lane >> 4;
    const int head = blockIdx.y;
    const int qRow = blockIdx.x * 128 + 16 * wave;
    const size_t ldq = QKVN;
    const float scale = 0.08838834764831845f;     // 1/sqrt(128)

    // Q fragments for this wave's 16 rows, all 128 dk, held in registers.
    FragB16 qf[4];
    const unsigned short* qptr = qkv + (size_t)(qRow + ln) * ldq + head * DHEAD;
#pragma unroll
    for (int f = 0; f < 4; ++f) {
        qf[f].q[0] = *(const u32x4*)(qptr + 32 * f + 8 * hh);
        qf[f].q[1] = *(const u32x4*)(qptr + 32 * f + 16 + 8 * hh);
    }

    const f32x8 zero = {0.f, 0.f, 0.f, 0.f, 0.f, 0.f, 0.f, 0.f};
    f32x8 o[8];
#pragma unroll
    for (int i = 0; i < 8; ++i) o[i] = zero;
    float mrow[8], lrow[8];
#pragma unroll
    for (int r = 0; r < 8; ++r) { mrow[r] = -1e30f; lrow[r] = 0.0f; }

    for (int kb = 0; kb < SEQ; kb += 64) {
        // K tile: 64 keys x 128 dk -- async DMA straight into LDS
#pragma unroll
        for (int c = 0; c < 4; ++c) {
            int id = t + c * 256;
            int row = id >> 4, cc = (id & 15) * 8;
            copy16_async(qkv + (size_t)(kb + row) * ldq + DMODEL + head * DHEAD + cc,
                         Kt + row * 136 + cc);
        }
        // V tile transposed: Vt[col][key] (overlaps the async K DMA)
#pragma unroll
        for (int c = 0; c < 4; ++c) {
            int id = t + c * 256;
            int row = id >> 4, cc = (id & 15) * 8;
            u32x4 v = *(const u32x4*)(qkv + (size_t)(kb + row) * ldq +
                                      2 * DMODEL + head * DHEAD + cc);
            const unsigned short* pv = (const unsigned short*)&v;
#pragma unroll
            for (int j = 0; j < 8; ++j) Vt[(cc + j) * 72 + row] = pv[j];
        }
        async_fence<0>();
        __syncthreads();

        // S = Q * K^T : 4 tiles of 16x16 over the 64 keys
        f32x8 s[4];
#pragma unroll
        for (int kt = 0; kt < 4; ++kt) {
            s[kt] = zero;
#pragma unroll
            for (int f = 0; f < 4; ++f) {
                // B[k=16(i>=8)+8hh+(i&7)][N=key 16kt+ln] = Kt[16kt+ln][32f + k]
                FragB16 b;
                b.q[0] = *(const u32x4*)(Kt + (16 * kt + ln) * 136 + 32 * f + 8 * hh);
                b.q[1] = *(const u32x4*)(Kt + (16 * kt + ln) * 136 + 32 * f + 16 + 8 * hh);
                s[kt] = __builtin_amdgcn_wmma_f32_16x16x32_bf16(
                    false, qf[f].v, false, b.v, (short)0, s[kt], false, false);
            }
            s[kt] = s[kt] * scale;
        }

        // Online softmax over this wave's 16 rows (8 per half-wave).
        float pr[4][8], corr[8];
#pragma unroll
        for (int r = 0; r < 8; ++r) {
            float mx = fmaxf(fmaxf(s[0][r], s[1][r]), fmaxf(s[2][r], s[3][r]));
#pragma unroll
            for (int off = 1; off < 16; off <<= 1)
                mx = fmaxf(mx, __shfl_xor(mx, off, 32));   // stays in half
            float mn = fmaxf(mrow[r], mx);
            float cr = __expf(mrow[r] - mn);
            float rs = 0.0f;
#pragma unroll
            for (int kt = 0; kt < 4; ++kt) {
                float p = __expf(s[kt][r] - mn);
                pr[kt][r] = p;
                rs += p;
            }
#pragma unroll
            for (int off = 1; off < 16; off <<= 1) rs += __shfl_xor(rs, off, 32);
            lrow[r] = lrow[r] * cr + rs;
            mrow[r] = mn;
            corr[r] = cr;
        }
#pragma unroll
        for (int n = 0; n < 8; ++n)
#pragma unroll
            for (int r = 0; r < 8; ++r) o[n][r] *= corr[r];

        // Re-layout P (D-layout -> A-layout) through wave-private LDS patch.
        unsigned short* P = Pst + wave * 16 * 72;
#pragma unroll
        for (int kt = 0; kt < 4; ++kt)
#pragma unroll
            for (int r = 0; r < 8; ++r)
                P[(8 * hh + r) * 72 + 16 * kt + ln] = f32_to_bf16u(pr[kt][r]);

        // O += P * V  (two K=32 steps over the 64 keys)
#pragma unroll
        for (int c = 0; c < 2; ++c) {
            FragB16 a;
            a.q[0] = *(const u32x4*)(P + ln * 72 + 32 * c + 8 * hh);
            a.q[1] = *(const u32x4*)(P + ln * 72 + 32 * c + 16 + 8 * hh);
#pragma unroll
            for (int n = 0; n < 8; ++n) {
                // B[k][N=col 16n+ln] = V[key 32c+16(i>=8)+8hh+(i&7)][col] = Vt[col][...]
                FragB16 b;
                b.q[0] = *(const u32x4*)(Vt + (16 * n + ln) * 72 + 32 * c + 8 * hh);
                b.q[1] = *(const u32x4*)(Vt + (16 * n + ln) * 72 + 32 * c + 16 + 8 * hh);
                o[n] = __builtin_amdgcn_wmma_f32_16x16x32_bf16(
                    false, a.v, false, b.v, (short)0, o[n], false, false);
            }
        }
        __syncthreads();
    }

    // Normalize and store to concat-heads layout [SEQ, H*128] (bf16).
#pragma unroll
    for (int n = 0; n < 8; ++n) {
        int col = head * DHEAD + 16 * n + ln;
#pragma unroll
        for (int r = 0; r < 8; ++r) {
            int row = qRow + 8 * hh + r;
            heads[(size_t)row * DMODEL + col] = f32_to_bf16u(o[n][r] / lrow[r]);
        }
    }
}

// ---------------------------------------------------------------------------
// Residual + LayerNorm over D=2048; one row per block (256 threads, 8/thread).
// Emits f32 (for later residual) and optional bf16 (for next GEMM).
// ---------------------------------------------------------------------------
__global__ __launch_bounds__(256) void resid_ln_kernel(
    const float* __restrict__ X, const float* __restrict__ Y,
    const float* __restrict__ g, const float* __restrict__ b,
    float* __restrict__ outF, unsigned short* __restrict__ outB) {
    const int row = blockIdx.x;
    const int t = threadIdx.x;
    const float* x = X + (size_t)row * DMODEL;
    const float* y = Y + (size_t)row * DMODEL;

    float v[8];
    float sum = 0.f, sq = 0.f;
#pragma unroll
    for (int i = 0; i < 8; ++i) {
        int c = t + i * 256;
        float f = x[c] + y[c];
        v[i] = f;
        sum += f;
        sq += f * f;
    }
#pragma unroll
    for (int off = 1; off < 32; off <<= 1) {
        sum += __shfl_xor(sum, off, 32);
        sq += __shfl_xor(sq, off, 32);
    }
    __shared__ float ssum[8], ssq[8];
    if ((t & 31) == 0) { ssum[t >> 5] = sum; ssq[t >> 5] = sq; }
    __syncthreads();
    float S = 0.f, Q = 0.f;
#pragma unroll
    for (int i = 0; i < 8; ++i) { S += ssum[i]; Q += ssq[i]; }
    const float inv = 1.0f / (float)DMODEL;
    float mean = S * inv;
    float var = Q * inv - mean * mean;
    float rstd = rsqrtf(var + 1e-5f);
#pragma unroll
    for (int i = 0; i < 8; ++i) {
        int c = t + i * 256;
        float f = (v[i] - mean) * rstd * g[c] + b[c];
        if (outF) outF[(size_t)row * DMODEL + c] = f;
        if (outB) outB[(size_t)row * DMODEL + c] = f32_to_bf16u(f);
    }
}

// ---------------------------------------------------------------------------
// Host launcher
// ---------------------------------------------------------------------------
extern "C" void kernel_launch(void* const* d_in, const int* in_sizes, int n_in,
                              void* d_out, int out_size, void* d_ws, size_t ws_size,
                              hipStream_t stream) {
    (void)in_sizes; (void)n_in; (void)out_size; (void)ws_size;
    const float* x      = (const float*)d_in[0];
    const float* wq     = (const float*)d_in[1];
    const float* wk     = (const float*)d_in[2];
    const float* wv     = (const float*)d_in[3];
    const float* w_proj = (const float*)d_in[4];
    const float* g1     = (const float*)d_in[5];
    const float* b1     = (const float*)d_in[6];
    const float* fc1_w  = (const float*)d_in[7];
    const float* fc1_b  = (const float*)d_in[8];
    const float* fc2_w  = (const float*)d_in[9];
    const float* fc2_b  = (const float*)d_in[10];
    const float* g2     = (const float*)d_in[11];
    const float* b2     = (const float*)d_in[12];
    float* out = (float*)d_out;

    char* ws = (char*)d_ws;
    size_t off = 0;
    auto alloc = [&](size_t bytes) {
        void* p = ws + off;
        off = (off + bytes + 255) & ~(size_t)255;
        return p;
    };
    unsigned short* xb     = (unsigned short*)alloc((size_t)SEQ * DMODEL * 2);
    unsigned short* wqkvb  = (unsigned short*)alloc((size_t)DMODEL * QKVN * 2);
    unsigned short* qkvb   = (unsigned short*)alloc((size_t)SEQ * QKVN * 2);
    unsigned short* wpb    = (unsigned short*)alloc((size_t)DMODEL * DMODEL * 2);
    unsigned short* headsb = (unsigned short*)alloc((size_t)SEQ * DMODEL * 2);
    float*          ao     = (float*)alloc((size_t)SEQ * DMODEL * 4);
    float*          hf     = (float*)alloc((size_t)SEQ * DMODEL * 4);
    unsigned short* hb     = (unsigned short*)alloc((size_t)SEQ * DMODEL * 2);
    unsigned short* fc1b   = (unsigned short*)alloc((size_t)DMODEL * DFF * 2);
    unsigned short* f1b    = (unsigned short*)alloc((size_t)SEQ * DFF * 2);
    unsigned short* fc2b   = (unsigned short*)alloc((size_t)DFF * DMODEL * 2);
    float*          ff     = (float*)alloc((size_t)SEQ * DMODEL * 4);

    // 1) precision conversion / weight packing
    cvt_bf16_kernel<<<(SEQ * DMODEL) / 256, 256, 0, stream>>>(x, xb, SEQ * DMODEL);
    pack_wqkv_kernel<<<(DMODEL * QKVN) / 256, 256, 0, stream>>>(wq, wk, wv, wqkvb);
    cvt_bf16_kernel<<<(DMODEL * DMODEL) / 256, 256, 0, stream>>>(w_proj, wpb, DMODEL * DMODEL);
    cvt_bf16_kernel<<<(DMODEL * DFF) / 256, 256, 0, stream>>>(fc1_w, fc1b, DMODEL * DFF);
    cvt_bf16_kernel<<<(DFF * DMODEL) / 256, 256, 0, stream>>>(fc2_w, fc2b, DFF * DMODEL);

    // 2) fused QKV projection: [S,2048] x [2048,6144] -> bf16
    gemm_bf16_wmma<false, false, true><<<dim3(QKVN / 128, SEQ / 128), 256, 0, stream>>>(
        xb, wqkvb, nullptr, nullptr, qkvb, SEQ, QKVN, DMODEL);

    // 3) flash attention per (q-block, head) -> concat heads bf16 [S,2048]
    attn_flash_kernel<<<dim3(SEQ / 128, NHEADS), 256, 0, stream>>>(qkvb, headsb);

    // 4) output projection -> f32
    gemm_bf16_wmma<false, false, false><<<dim3(DMODEL / 128, SEQ / 128), 256, 0, stream>>>(
        headsb, wpb, nullptr, ao, nullptr, SEQ, DMODEL, DMODEL);

    // 5) h = LN(x + attn_out)
    resid_ln_kernel<<<SEQ, 256, 0, stream>>>(x, ao, g1, b1, hf, hb);

    // 6) f1 = relu(h @ fc1 + b) -> bf16
    gemm_bf16_wmma<true, true, true><<<dim3(DFF / 128, SEQ / 128), 256, 0, stream>>>(
        hb, fc1b, fc1_b, nullptr, f1b, SEQ, DFF, DMODEL);

    // 7) ff = f1 @ fc2 + b -> f32
    gemm_bf16_wmma<true, false, false><<<dim3(DMODEL / 128, SEQ / 128), 256, 0, stream>>>(
        f1b, fc2b, fc2_b, ff, nullptr, SEQ, DMODEL, DFF);

    // 8) out = LN(h + ff)
    resid_ln_kernel<<<SEQ, 256, 0, stream>>>(hf, ff, g2, b2, out, nullptr);
}
